// Model_31086973288698
// MI455X (gfx1250) — matrix operand order repaired and checked by reading, not compiled, not run on hardware
//
#include <hip/hip_runtime.h>
#include <math.h>

// Problem constants (match reference)
#define NI_  2048
#define H_   1024
#define B_   256
#define T_   256
#define G4_  4096   // 4*H

typedef __attribute__((ext_vector_type(16))) __bf16 v16bf;
typedef __attribute__((ext_vector_type(8)))  __bf16 v8bf;
typedef __attribute__((ext_vector_type(8)))  float  v8f;
typedef __attribute__((ext_vector_type(4)))  unsigned int v4u;
typedef __attribute__((ext_vector_type(8)))  int v8i_;
typedef __attribute__((ext_vector_type(4)))  int v4i_;

#if defined(__has_builtin)
# if __has_builtin(__builtin_amdgcn_tensor_load_to_lds) && __has_builtin(__builtin_amdgcn_s_wait_tensorcnt)
#  define USE_TDM 1
# endif
#endif
#ifndef USE_TDM
# define USE_TDM 0
#endif

__device__ __forceinline__ unsigned short f2bf(float f) {
    unsigned int u = __float_as_uint(f);
    u = (u + 0x7FFFu + ((u >> 16) & 1u)) >> 16;   // round-to-nearest-even
    return (unsigned short)u;
}

__device__ __forceinline__ float sigm(float x) {
    return 1.0f / (1.0f + __expf(-x));
}

__device__ __forceinline__ unsigned int mixhash(unsigned int x) {
    x ^= x >> 16; x *= 0x7feb352dU;
    x ^= x >> 15; x *= 0x846ca68bU;
    x ^= x >> 16; return x;
}

// ---------------------------------------------------------------------------
__global__ __launch_bounds__(256) void zero_mem(unsigned int* __restrict__ p, int n) {
    int i = blockIdx.x * 256 + threadIdx.x;
    if (i < n) p[i] = 0u;
}

// ---------------------------------------------------------------------------
__global__ __launch_bounds__(256) void compute_lens(const int* __restrict__ items,
                                                    int* __restrict__ lens) {
    int b = threadIdx.x;
    int cnt = 0;
    for (int t = 0; t < T_; ++t) cnt += (items[b * T_ + t] != 0);
    lens[b] = cnt;
}

// ---------------------------------------------------------------------------
// Repack Whh (H x 4H row-major f32) into bf16 WMMA B-operand tile layout:
//   out[((nb*32 + kb)*32 + lane)*16 + e]  where
//   n = nb*16 + (lane&15),  k = kb*32 + (e>=8 ? 16 : 0) + (lane>=16 ? 8 : 0) + (e&7)
__global__ __launch_bounds__(256) void pack_whh(const float* __restrict__ Whh,
                                                unsigned short* __restrict__ out) {
    int idx  = blockIdx.x * 256 + threadIdx.x;      // [0, 4M)
    int e    = idx & 15;
    int lane = (idx >> 4) & 31;
    int kb   = (idx >> 9) & 31;
    int nb   = idx >> 14;
    int k = kb * 32 + ((e >> 3) << 4) + ((lane >> 4) << 3) + (e & 7);
    int n = nb * 16 + (lane & 15);
    out[idx] = f2bf(Whh[k * G4_ + n]);
}

// ---------------------------------------------------------------------------
// One fused LSTM time step. Grid: 128 blocks x 256 threads.
// Block = one btile (16 batch rows); its 8 waves cover 8 consecutive ntiles.
// The block's shared 16x1024 bf16 h tile is DMA'd into LDS by the TDM once,
// then each wave runs K=1024 with 4 gate accumulators:
// 128 v_wmma_f32_16x16x32_bf16 per wave, A from ds_load_b128, B from
// register-double-buffered global_load_b128 streams out of L2.
__global__ __launch_bounds__(256, 1) void lstm_step(
    const float* __restrict__ Wt,            // (2*NI, 4H) f32
    const float* __restrict__ bias,          // (4H)
    const unsigned short* __restrict__ WhhT, // packed bf16 tiles
    const int* __restrict__ items,           // (B,T)
    const int* __restrict__ actions,         // (B,T)
    const int* __restrict__ lens,            // (B)
    float* __restrict__ h,                   // (B,H) f32 state (in-place)
    float* __restrict__ c,                   // (B,H) f32 state (in-place)
    const unsigned short* __restrict__ hbf_rd, // (B,H) bf16 mirror, read
    unsigned short* __restrict__ hbf_wr,       // (B,H) bf16 mirror, write
    int t)
{
    __shared__ __align__(16) unsigned short shA[16 * H_];   // 32 KB

    const int lane  = threadIdx.x & 31;
    const int warp  = threadIdx.x >> 5;
    const int btile = blockIdx.x >> 3;               // 0..15
    const int ntile = (blockIdx.x & 7) * 8 + warp;   // 0..63

#if USE_TDM
    if (warp == 0) {
        // 2D Tensor DMA descriptor (D#): 16 rows x 1024 cols of bf16,
        // row stride 1024 elements, contiguous into LDS.
        unsigned long long ga =
            (unsigned long long)(size_t)(hbf_rd + (size_t)btile * 16 * H_);
        unsigned ldsb = (unsigned)(size_t)(&shA[0]);
        v4u g0;
        g0.x = 1u;                                           // count=1 (valid)
        g0.y = ldsb;                                         // lds_addr
        g0.z = (unsigned)ga;                                 // global_addr[31:0]
        g0.w = (unsigned)((ga >> 32) & 0x01FFFFFFu) | (2u << 30); // [56:32] | type=2
        v8i_ g1;
        g1[0] = 0x00010000;                    // data_size = 2 bytes, no masks
        g1[1] = (int)((H_ & 0xFFFF) << 16);    // tensor_dim0 = 1024 (lo16)
        g1[2] = (int)(16 << 16);               // tensor_dim0 hi=0 | tensor_dim1=16
        g1[3] = (int)((H_ & 0xFFFF) << 16);    // tensor_dim1 hi=0 | tile_dim0=1024
        g1[4] = 16;                            // tile_dim1=16, tile_dim2=0
        g1[5] = H_;                            // tensor_dim0_stride = 1024 (lo32)
        g1[6] = 0;                             // stride hi, dim1_stride lo
        g1[7] = 0;
        v4i_ z4 = (v4i_)0;                     // groups 2/3 unused (2D tensor)
        v8i_ z8 = (v8i_)0;                     // trailing group (6-arg variant)
        __builtin_amdgcn_tensor_load_to_lds(g0, g1, z4, z4, z8, 0);
        __builtin_amdgcn_s_wait_tensorcnt(0);
    }
    __syncthreads();
#else
    // Fallback cooperative copy: 32 KB in 16B chunks.
    {
        const float4* src = (const float4*)(hbf_rd + (size_t)btile * 16 * H_);
        float4* dst = (float4*)shA;
        for (int i = threadIdx.x; i < (16 * H_ * 2) / 16; i += 256) dst[i] = src[i];
    }
    __syncthreads();
#endif

    // A-operand (16x32 bf16): lane -> row M = lane&15; K-halves by lane group.
    const int koff = (lane >> 4) << 3;                 // 0 or 8
    const unsigned short* aptr = shA + (lane & 15) * H_ + koff;

    // B streams: one packed tile run per gate (gate g at columns g*H + [0,H)).
    const unsigned short* bp0 = WhhT + (ntile      ) * 16384 + lane * 16;
    const unsigned short* bp1 = WhhT + (ntile +  64) * 16384 + lane * 16;
    const unsigned short* bp2 = WhhT + (ntile + 128) * 16384 + lane * 16;
    const unsigned short* bp3 = WhhT + (ntile + 192) * 16384 + lane * 16;

    v8f acc0 = {}, acc1 = {}, acc2 = {}, acc3 = {};

    // Software pipeline: prefetch kb+1 operands while kb's 4 WMMAs issue.
    v16bf ac, b0c, b1c, b2c, b3c;
    ((v8bf*)&ac)[0] = *(const v8bf*)(aptr);
    ((v8bf*)&ac)[1] = *(const v8bf*)(aptr + 16);
    b0c = *(const v16bf*)(bp0);
    b1c = *(const v16bf*)(bp1);
    b2c = *(const v16bf*)(bp2);
    b3c = *(const v16bf*)(bp3);

#pragma unroll
    for (int kb = 0; kb < 32; ++kb) {
        v16bf an = ac, b0n = b0c, b1n = b1c, b2n = b2c, b3n = b3c;
        if (kb < 31) {
            const int ao = (kb + 1) * 32;
            const int bo = (kb + 1) * 512;
            ((v8bf*)&an)[0] = *(const v8bf*)(aptr + ao);
            ((v8bf*)&an)[1] = *(const v8bf*)(aptr + ao + 16);
            b0n = *(const v16bf*)(bp0 + bo);
            b1n = *(const v16bf*)(bp1 + bo);
            b2n = *(const v16bf*)(bp2 + bo);
            b3n = *(const v16bf*)(bp3 + bo);
        }
        acc0 = __builtin_amdgcn_wmma_f32_16x16x32_bf16(false, ac, false, b0c, (short)0, acc0, false, false);
        acc1 = __builtin_amdgcn_wmma_f32_16x16x32_bf16(false, ac, false, b1c, (short)0, acc1, false, false);
        acc2 = __builtin_amdgcn_wmma_f32_16x16x32_bf16(false, ac, false, b2c, (short)0, acc2, false, false);
        acc3 = __builtin_amdgcn_wmma_f32_16x16x32_bf16(false, ac, false, b3c, (short)0, acc3, false, false);
        ac = an; b0c = b0n; b1c = b1n; b2c = b2n; b3c = b3n;
    }

    float a0[8], a1[8], a2[8], a3[8];
#pragma unroll
    for (int r = 0; r < 8; ++r) { a0[r] = acc0[r]; a1[r] = acc1[r]; a2[r] = acc2[r]; a3[r] = acc3[r]; }

    // D layout: VGPR r -> row M = r + (lane>=16 ? 8 : 0), col N = lane&15
    const int ncol  = ntile * 16 + (lane & 15);
    const int mbase = btile * 16 + ((lane >> 4) << 3);

    const float bi = bias[ncol];
    const float bf = bias[ncol + H_];
    const float bg = bias[ncol + 2 * H_];
    const float bo = bias[ncol + 3 * H_];

#pragma unroll
    for (int r = 0; r < 8; ++r) {
        const int gb = mbase + r;
        const int it = items[gb * T_ + t];
        const int ac_ = actions[gb * T_ + t];
        const bool valid = (t < lens[gb]);

        const float* wi = Wt + (size_t)it * G4_;
        const float* wa = Wt + (size_t)(NI_ + ac_) * G4_;

        const float gi = a0[r] + wi[ncol]          + wa[ncol]          + bi;
        const float gf = a1[r] + wi[ncol +     H_] + wa[ncol +     H_] + bf;
        const float gg = a2[r] + wi[ncol + 2 * H_] + wa[ncol + 2 * H_] + bg;
        const float go = a3[r] + wi[ncol + 3 * H_] + wa[ncol + 3 * H_] + bo;

        const int idx = gb * H_ + ncol;
        const float hold = h[idx];
        const float cold = c[idx];

        const float iv = sigm(gi), fv = sigm(gf), ov = sigm(go);
        const float cn = fv * cold + iv * tanhf(gg);
        const float hn = ov * tanhf(cn);

        if (valid) { h[idx] = hn; c[idx] = cn; }
        hbf_wr[idx] = f2bf(valid ? hn : hold);
    }
}

// ---------------------------------------------------------------------------
// Dropout (hash-based 0.5 mask approximating the fixed-key bernoulli),
// FC head (2048 x 2), log-softmax, mean NLL.
__global__ __launch_bounds__(256) void final_loss(
    const float* __restrict__ hf, const float* __restrict__ hb,
    const int* __restrict__ targets,
    const float* __restrict__ Wfc, const float* __restrict__ bfc,
    float* __restrict__ out)
{
    __shared__ float red[256];
    const int b = threadIdx.x;
    float l0 = bfc[0], l1 = bfc[1];

    for (int j = 0; j < H_; ++j) {
        const unsigned keep = mixhash((unsigned)(b * H_ + j)) & 1u;
        const float v = keep ? hf[b * H_ + j] * 2.0f : 0.0f;
        l0 += v * Wfc[j * 2 + 0];
        l1 += v * Wfc[j * 2 + 1];
    }
    for (int j = 0; j < H_; ++j) {
        const unsigned keep = mixhash((unsigned)(B_ * H_ + b * H_ + j)) & 1u;
        const float v = keep ? hb[b * H_ + j] * 2.0f : 0.0f;
        l0 += v * Wfc[(H_ + j) * 2 + 0];
        l1 += v * Wfc[(H_ + j) * 2 + 1];
    }

    const float m   = fmaxf(l0, l1);
    const float lse = m + __logf(__expf(l0 - m) + __expf(l1 - m));
    const int   tgt = targets[b];
    red[b] = -(((tgt == 0) ? l0 : l1) - lse);
    __syncthreads();
    for (int s = 128; s > 0; s >>= 1) {
        if (b < s) red[b] += red[b + s];
        __syncthreads();
    }
    if (b == 0) out[0] = red[0] / (float)B_;
}

// ---------------------------------------------------------------------------
extern "C" void kernel_launch(void* const* d_in, const int* in_sizes, int n_in,
                              void* d_out, int out_size, void* d_ws, size_t ws_size,
                              hipStream_t stream) {
    (void)in_sizes; (void)n_in; (void)out_size; (void)ws_size;

    const int*   items   = (const int*)  d_in[0];
    const int*   actions = (const int*)  d_in[1];
    const int*   targets = (const int*)  d_in[2];
    const float* Wt_f    = (const float*)d_in[3];
    const float* Whh_f   = (const float*)d_in[4];
    const float* b_f     = (const float*)d_in[5];
    const float* Wt_b    = (const float*)d_in[6];
    const float* Whh_b   = (const float*)d_in[7];
    const float* b_b     = (const float*)d_in[8];
    const float* Wfc     = (const float*)d_in[9];
    const float* bfc     = (const float*)d_in[10];

    char* ws = (char*)d_ws;
    unsigned short* WhhT_f = (unsigned short*)(ws + 0);         // 8 MB
    unsigned short* WhhT_b = (unsigned short*)(ws + 8388608);   // 8 MB
    float* h_f  = (float*)(ws + 16777216);                      // 1 MB each
    float* c_f  = (float*)(ws + 17825792);
    float* h_b  = (float*)(ws + 18874368);
    float* c_b  = (float*)(ws + 19922944);
    unsigned short* hbfA_f = (unsigned short*)(ws + 20971520);  // 512 KB each
    unsigned short* hbfB_f = (unsigned short*)(ws + 21495808);
    unsigned short* hbfA_b = (unsigned short*)(ws + 22020096);
    unsigned short* hbfB_b = (unsigned short*)(ws + 22544384);
    int* lens = (int*)(ws + 23068672);

    // Zero all recurrent state (h, c, both bf16 mirrors, both dirs): 6 MB.
    zero_mem<<<6144, 256, 0, stream>>>((unsigned int*)(ws + 16777216), 1572864);
    compute_lens<<<1, 256, 0, stream>>>(items, lens);
    pack_whh<<<16384, 256, 0, stream>>>(Whh_f, WhhT_f);
    pack_whh<<<16384, 256, 0, stream>>>(Whh_b, WhhT_b);

    // Host loop over time provides the inter-step grid barrier; bf16 h mirror
    // is double-buffered (all waves read all of h while producing the next h).
    for (int s = 0; s < T_; ++s) {
        const unsigned short* rdF = (s & 1) ? hbfB_f : hbfA_f;
        unsigned short*       wrF = (s & 1) ? hbfA_f : hbfB_f;
        lstm_step<<<128, 256, 0, stream>>>(Wt_f, b_f, WhhT_f, items, actions, lens,
                                           h_f, c_f, rdF, wrF, s);
        const unsigned short* rdB = (s & 1) ? hbfB_b : hbfA_b;
        unsigned short*       wrB = (s & 1) ? hbfA_b : hbfB_b;
        lstm_step<<<128, 256, 0, stream>>>(Wt_b, b_b, WhhT_b, items, actions, lens,
                                           h_b, c_b, rdB, wrB, T_ - 1 - s);
    }

    final_loss<<<1, 256, 0, stream>>>(h_f, h_b, targets, Wfc, bfc, (float*)d_out);
}
